// INT8Linear_11888469475634
// MI455X (gfx1250) — compile-verified, matrix operand order checked by reading, not run on hardware
//
#include <hip/hip_runtime.h>
#include <stdint.h>

typedef _Float16 v16h __attribute__((ext_vector_type(16)));
typedef _Float16 v8h  __attribute__((ext_vector_type(8)));
typedef _Float16 v4h  __attribute__((ext_vector_type(4)));
typedef float    v8f  __attribute__((ext_vector_type(8)));

#define DIN   4096
#define DOUT  11008
#define MTOT  (8 * 2048)

#define KT     64
#define MTILE  128
#define NTILE  128
#define LDA    72   // padded LDS row stride (f16): 36 DWORDs -> conflict-free b128 reads
#define LDB    72

__global__ __launch_bounds__(256)
void int8_linear_wmma(const float* __restrict__ x,
                      const int8_t* __restrict__ w,
                      const float* __restrict__ scale,
                      float* __restrict__ out)
{
    __shared__ alignas(16) _Float16 sA[2][MTILE * LDA];
    __shared__ alignas(16) _Float16 sB[2][NTILE * LDB];

    const int tid    = threadIdx.x;
    const int lane   = tid & 31;
    const int laneLo = lane & 15;
    const int laneHi = lane >> 4;
    const int wave   = tid >> 5;      // 0..7
    const int waveM  = wave >> 1;     // 0..3 (M direction)
    const int waveN  = wave & 1;      // 0..1 (N direction)

    const int nBase = blockIdx.x * NTILE;   // N fastest -> L2 reuse of X stripe
    const int mBase = blockIdx.y * MTILE;

    // A loader: 16 threads/row, one float4 each (256B coalesced per row)
    const int aRow = tid >> 4;        // 0..15, +16 per pass
    const int aCol = (tid & 15) * 4;
    // B loader: 8 threads/row, 8 int8 bytes each
    const int bRow = tid >> 3;        // 0..31, +32 per pass
    const int bCol = (tid & 7) * 8;

    v8f acc[2][4] = {};               // 64 VGPRs of f32 accumulators

    // ---- pipeline staging registers (tile k+1 in flight during compute of k) ----
    float4 aReg[8];
    uint2  bReg[4];

    auto gload = [&](int k0) {
        #pragma unroll
        for (int p = 0; p < 8; ++p)
            aReg[p] = *(const float4*)(x + (size_t)(mBase + aRow + p * 16) * DIN + k0 + aCol);
        #pragma unroll
        for (int p = 0; p < 4; ++p)
            bReg[p] = *(const uint2*)(w + (size_t)(nBase + bRow + p * 32) * DIN + k0 + bCol);
    };

    auto stage = [&](int buf) {
        #pragma unroll
        for (int p = 0; p < 8; ++p) {
            const float4 f = aReg[p];
            v4h h;
            h.x = (_Float16)f.x; h.y = (_Float16)f.y;
            h.z = (_Float16)f.z; h.w = (_Float16)f.w;
            *(v4h*)(&sA[buf][(aRow + p * 16) * LDA + aCol]) = h;
        }
        #pragma unroll
        for (int p = 0; p < 4; ++p) {
            const uint2 q = bReg[p];
            v8h h;
            #pragma unroll
            for (int b = 0; b < 4; ++b) {
                h[b]     = (_Float16)(int)(int8_t)(q.x >> (8 * b));   // exact int8 -> f16
                h[b + 4] = (_Float16)(int)(int8_t)(q.y >> (8 * b));
            }
            *(v8h*)(&sB[buf][(bRow + p * 32) * LDB + bCol]) = h;
        }
    };

    auto compute = [&](int buf) {
        #pragma unroll
        for (int kk = 0; kk < KT; kk += 32) {
            v16h aF[2], bF[4];
            #pragma unroll
            for (int fm = 0; fm < 2; ++fm) {
                const _Float16* p = &sA[buf][(waveM * 32 + fm * 16 + laneLo) * LDA + kk + laneHi * 8];
                *(v8h*)&aF[fm]       = *(const v8h*)p;         // K 0..7   (or 8..15)
                *((v8h*)&aF[fm] + 1) = *(const v8h*)(p + 16);  // K 16..23 (or 24..31)
            }
            #pragma unroll
            for (int fn = 0; fn < 4; ++fn) {
                const _Float16* p = &sB[buf][(waveN * 64 + fn * 16 + laneLo) * LDB + kk + laneHi * 8];
                *(v8h*)&bF[fn]       = *(const v8h*)p;
                *((v8h*)&bF[fn] + 1) = *(const v8h*)(p + 16);
            }
            #pragma unroll
            for (int fm = 0; fm < 2; ++fm)
                #pragma unroll
                for (int fn = 0; fn < 4; ++fn)
                    acc[fm][fn] = __builtin_amdgcn_wmma_f32_16x16x32_f16(
                        false, aF[fm], false, bF[fn],
                        (short)0, acc[fm][fn], false, false);
        }
    };

    // ---- software pipeline: load k+1 while computing k ----
    gload(0);
    stage(0);
    __syncthreads();

    int cur = 0;
    for (int k0 = KT; k0 < DIN; k0 += KT) {
        gload(k0);                       // long-latency HBM loads issued first
        if (k0 + KT < DIN) {             // pull tile k+2 toward L2 meanwhile
            __builtin_prefetch(x + (size_t)(mBase + aRow) * DIN + k0 + KT + aCol, 0, 1);
            __builtin_prefetch(w + (size_t)(nBase + bRow) * DIN + k0 + KT + bCol, 0, 1);
        }
        compute(cur);                    // 16 WMMAs + 24 ds_loads hide the loads
        stage(cur ^ 1);                  // waits on loadcnt only here
        __syncthreads();
        cur ^= 1;
    }
    compute(cur);

    // ---- epilogue: scale, non-temporal store (don't evict W/X from L2) ----
    #pragma unroll
    for (int fm = 0; fm < 2; ++fm) {
        const int row0 = mBase + waveM * 32 + fm * 16 + laneHi * 8;
        #pragma unroll
        for (int fn = 0; fn < 4; ++fn) {
            const int col0 = nBase + waveN * 64 + fn * 16;
            const float s = scale[col0 + laneLo];
            float* o = out + (size_t)row0 * DOUT + col0 + laneLo;
            #pragma unroll
            for (int i = 0; i < 8; ++i) {
                const float v = acc[fm][fn][i] * s;
                __builtin_nontemporal_store(v, o + (size_t)i * DOUT);
            }
        }
    }
}

extern "C" void kernel_launch(void* const* d_in, const int* in_sizes, int n_in,
                              void* d_out, int out_size, void* d_ws, size_t ws_size,
                              hipStream_t stream) {
    const float*  x     = (const float*)d_in[0];
    const int8_t* w     = (const int8_t*)d_in[1];
    const float*  scale = (const float*)d_in[2];
    float*        out   = (float*)d_out;

    dim3 grid(DOUT / NTILE, MTOT / MTILE);   // (86, 128); N fastest for L2 reuse
    dim3 block(256);
    int8_linear_wmma<<<grid, block, 0, stream>>>(x, w, scale, out);
}
